// VectorQuantizer_86466281603560
// MI455X (gfx1250) — matrix operand order repaired and checked by reading, not compiled, not run on hardware
//
#include <hip/hip_runtime.h>

typedef __attribute__((ext_vector_type(16))) __bf16 v16bf;
typedef __attribute__((ext_vector_type(8)))  __bf16 v8bf;
typedef __attribute__((ext_vector_type(8)))  float  v8f;

#define NUM_CODES 4096
#define CODE_DIM  256
#define BATCH     65536

// ---------------------------------------------------------------------------
// Kernel 1: codebook prep. One wave per codebook row (8 rows / 256-thr block).
// Writes bf16 copy of the codebook and per-code squared norms.
// ---------------------------------------------------------------------------
__global__ void __launch_bounds__(256)
vq_prep(const float* __restrict__ cb,
        __bf16* __restrict__ cb16,
        float* __restrict__ cn2) {
  const int wave = threadIdx.x >> 5;
  const int lane = threadIdx.x & 31;
  const int row  = blockIdx.x * 8 + wave;

  const float* src = cb + (size_t)row * CODE_DIM + lane * 8;
  float4 f0 = *reinterpret_cast<const float4*>(src);
  float4 f1 = *reinterpret_cast<const float4*>(src + 4);

  float s = f0.x * f0.x + f0.y * f0.y + f0.z * f0.z + f0.w * f0.w +
            f1.x * f1.x + f1.y * f1.y + f1.z * f1.z + f1.w * f1.w;

  v8bf h;
  h[0] = (__bf16)f0.x; h[1] = (__bf16)f0.y; h[2] = (__bf16)f0.z; h[3] = (__bf16)f0.w;
  h[4] = (__bf16)f1.x; h[5] = (__bf16)f1.y; h[6] = (__bf16)f1.z; h[7] = (__bf16)f1.w;
  *reinterpret_cast<v8bf*>(cb16 + (size_t)row * CODE_DIM + lane * 8) = h;

#pragma unroll
  for (int off = 16; off > 0; off >>= 1) s += __shfl_xor(s, off, 32);
  if (lane == 0) cn2[row] = s;
}

// ---------------------------------------------------------------------------
// Kernel 2: WMMA-based argmin over codes.
// Block = 256 threads = 8 waves; each wave owns a 16-row z tile (M=16).
// A = (-2*z) 16x32 bf16 fragments (K=256 -> 8 fragments, held in VGPRs for the
// whole kernel). B = codebook^T tiles, loaded straight from the row-major bf16
// codebook (B layout: lane n%16 holds code row n with contiguous K).
// WMMA accumulator is seeded with ||c_n||^2 broadcast, so the chain computes
// score[m][n] = ||c_n||^2 - 2 z_m . c_n directly.
// ---------------------------------------------------------------------------
__global__ void __launch_bounds__(256)
vq_argmin(const float* __restrict__ z,
          const __bf16* __restrict__ cb16,
          const float* __restrict__ cn2,
          int* __restrict__ idx_ws,
          float* __restrict__ out_idx_f) {
  const int wave = threadIdx.x >> 5;
  const int lane = threadIdx.x & 31;
  const int lrow = lane & 15;  // A: row M; B: column n (= code within tile)
  const int hi   = lane >> 4;  // which K-half this lane covers
  const int rowbase = blockIdx.x * 128 + wave * 16;

  // ---- Load z tile row (rowbase+lrow), convert to bf16 scaled by -2 ----
  // A 16x32 bf16 layout: lanes 0-15 hold K = 32c+[0..15],
  // lanes 16-31 hold K = 32c+[16..31]  (elements ascending).
  const float* zr = z + (size_t)(rowbase + lrow) * CODE_DIM;
  v16bf a[8];
#pragma unroll
  for (int c = 0; c < 8; ++c) {
    const int k0 = 32 * c + 16 * hi;
    float4 p0 = *reinterpret_cast<const float4*>(zr + k0);
    float4 p1 = *reinterpret_cast<const float4*>(zr + k0 + 4);
    float4 q0 = *reinterpret_cast<const float4*>(zr + k0 + 8);
    float4 q1 = *reinterpret_cast<const float4*>(zr + k0 + 12);
    a[c][0]  = (__bf16)(-2.0f * p0.x); a[c][1]  = (__bf16)(-2.0f * p0.y);
    a[c][2]  = (__bf16)(-2.0f * p0.z); a[c][3]  = (__bf16)(-2.0f * p0.w);
    a[c][4]  = (__bf16)(-2.0f * p1.x); a[c][5]  = (__bf16)(-2.0f * p1.y);
    a[c][6]  = (__bf16)(-2.0f * p1.z); a[c][7]  = (__bf16)(-2.0f * p1.w);
    a[c][8]  = (__bf16)(-2.0f * q0.x); a[c][9]  = (__bf16)(-2.0f * q0.y);
    a[c][10] = (__bf16)(-2.0f * q0.z); a[c][11] = (__bf16)(-2.0f * q0.w);
    a[c][12] = (__bf16)(-2.0f * q1.x); a[c][13] = (__bf16)(-2.0f * q1.y);
    a[c][14] = (__bf16)(-2.0f * q1.z); a[c][15] = (__bf16)(-2.0f * q1.w);
  }

  // B fragment base for this lane: code row (nt*16 + lrow), K-half = hi.
  const __bf16* bbase = cb16 + (size_t)lrow * CODE_DIM + 16 * hi;

  float minv[8];
  int   mini[8];
#pragma unroll
  for (int j = 0; j < 8; ++j) { minv[j] = 3.4e38f; mini[j] = 0; }

  for (int nt = 0; nt < NUM_CODES / 16; ++nt) {
    const __bf16* bp = bbase + (size_t)nt * 16 * CODE_DIM;
    const float c2 = cn2[nt * 16 + lrow];

    v8f acc0, acc1;
#pragma unroll
    for (int j = 0; j < 8; ++j) { acc0[j] = c2; acc1[j] = 0.0f; }

#pragma unroll
    for (int c = 0; c < 8; c += 2) {
      v16bf b0 = *reinterpret_cast<const v16bf*>(bp + 32 * c);
      v16bf b1 = *reinterpret_cast<const v16bf*>(bp + 32 * (c + 1));
      acc0 = __builtin_amdgcn_wmma_f32_16x16x32_bf16(
          false, a[c], false, b0, (short)0, acc0, false, false);
      acc1 = __builtin_amdgcn_wmma_f32_16x16x32_bf16(
          false, a[c + 1], false, b1, (short)0, acc1, false, false);
    }

    const int n = nt * 16 + lrow;
#pragma unroll
    for (int j = 0; j < 8; ++j) {
      float s = acc0[j] + acc1[j];
      if (s < minv[j]) { minv[j] = s; mini[j] = n; }
    }
  }

  // Cross-lane argmin over the 16 lanes that share each M row.
  // Lanes 0-15 hold M = 0..7 (VGPR j -> M=j), lanes 16-31 hold M = 8..15.
#pragma unroll
  for (int j = 0; j < 8; ++j) {
    float v = minv[j];
    int   i = mini[j];
#pragma unroll
    for (int off = 1; off < 16; off <<= 1) {
      float ov = __shfl_xor(v, off, 32);
      int   oi = __shfl_xor(i, off, 32);
      if (ov < v || (ov == v && oi < i)) { v = ov; i = oi; }
    }
    minv[j] = v; mini[j] = i;
  }

  if (lrow == 0) {
    const int mbase = rowbase + hi * 8;
#pragma unroll
    for (int j = 0; j < 8; ++j) {
      idx_ws[mbase + j]    = mini[j];
      out_idx_f[mbase + j] = (float)mini[j];
    }
  }
}

// ---------------------------------------------------------------------------
// Kernel 3: gather z_q = codebook[idx] (full f32) + per-block loss partials.
// One wave per z row; 8 rows per block.
// ---------------------------------------------------------------------------
__global__ void __launch_bounds__(256)
vq_gather(const float* __restrict__ z,
          const float* __restrict__ cb,
          const int* __restrict__ idx_ws,
          float* __restrict__ zq,
          float* __restrict__ partials) {
  __shared__ float sdata[8];
  const int wave = threadIdx.x >> 5;
  const int lane = threadIdx.x & 31;
  const int row  = blockIdx.x * 8 + wave;
  const int code = idx_ws[row];

  const float* c  = cb + (size_t)code * CODE_DIM + lane * 8;
  const float* zr = z  + (size_t)row  * CODE_DIM + lane * 8;
  float*       o  = zq + (size_t)row  * CODE_DIM + lane * 8;

  float4 c0 = *reinterpret_cast<const float4*>(c);
  float4 c1 = *reinterpret_cast<const float4*>(c + 4);
  float4 z0 = *reinterpret_cast<const float4*>(zr);
  float4 z1 = *reinterpret_cast<const float4*>(zr + 4);

  *reinterpret_cast<float4*>(o)     = c0;
  *reinterpret_cast<float4*>(o + 4) = c1;

  float dx, s = 0.0f;
  dx = c0.x - z0.x; s += dx * dx;  dx = c0.y - z0.y; s += dx * dx;
  dx = c0.z - z0.z; s += dx * dx;  dx = c0.w - z0.w; s += dx * dx;
  dx = c1.x - z1.x; s += dx * dx;  dx = c1.y - z1.y; s += dx * dx;
  dx = c1.z - z1.z; s += dx * dx;  dx = c1.w - z1.w; s += dx * dx;

#pragma unroll
  for (int off = 16; off > 0; off >>= 1) s += __shfl_xor(s, off, 32);
  if (lane == 0) sdata[wave] = s;
  __syncthreads();
  if (threadIdx.x == 0) {
    float t = 0.0f;
#pragma unroll
    for (int w = 0; w < 8; ++w) t += sdata[w];
    partials[blockIdx.x] = t;
  }
}

// ---------------------------------------------------------------------------
// Kernel 4: final deterministic loss reduction over 8192 partials.
// loss = (COMMITMENT_COST + 1) * mean((z_q - z)^2) = 1.25 * mse
// ---------------------------------------------------------------------------
__global__ void __launch_bounds__(256)
vq_loss(const float* __restrict__ partials, float* __restrict__ loss_out) {
  __shared__ float sdata[256];
  float s = 0.0f;
  for (int i = threadIdx.x; i < BATCH / 8; i += 256) s += partials[i];
  sdata[threadIdx.x] = s;
  __syncthreads();
  for (int st = 128; st > 0; st >>= 1) {
    if ((int)threadIdx.x < st) sdata[threadIdx.x] += sdata[threadIdx.x + st];
    __syncthreads();
  }
  if (threadIdx.x == 0)
    *loss_out = 1.25f * sdata[0] / ((float)BATCH * (float)CODE_DIM);
}

// ---------------------------------------------------------------------------
extern "C" void kernel_launch(void* const* d_in, const int* in_sizes, int n_in,
                              void* d_out, int out_size, void* d_ws, size_t ws_size,
                              hipStream_t stream) {
  const float* z  = (const float*)d_in[0];        // [BATCH, CODE_DIM]
  const float* cb = (const float*)d_in[1];        // [NUM_CODES, CODE_DIM]
  float* out = (float*)d_out;                     // zq | indices(as f32) | loss

  // Workspace layout
  char* ws = (char*)d_ws;
  __bf16* cb16     = (__bf16*)ws;                                   // 2 MB
  float*  cn2      = (float*)(ws + (size_t)NUM_CODES * CODE_DIM * 2);
  int*    idx_ws   = (int*)((char*)cn2 + NUM_CODES * sizeof(float));
  float*  partials = (float*)((char*)idx_ws + BATCH * sizeof(int)); // 8192 f32

  float* out_zq    = out;
  float* out_idx_f = out + (size_t)BATCH * CODE_DIM;
  float* out_loss  = out_idx_f + BATCH;

  vq_prep  <<<NUM_CODES / 8, 256, 0, stream>>>(cb, cb16, cn2);
  vq_argmin<<<BATCH / 128,   256, 0, stream>>>(z, cb16, cn2, idx_ws, out_idx_f);
  vq_gather<<<BATCH / 8,     256, 0, stream>>>(z, cb, idx_ws, out_zq, partials);
  vq_loss  <<<1,             256, 0, stream>>>(partials, out_loss);
}
